// PatchLocalPoolPointnetLatent_82660940578890
// MI455X (gfx1250) — matrix operand order repaired and checked by reading, not compiled, not run on hardware
//
#include <hip/hip_runtime.h>
#include <hip/hip_bf16.h>

#define HID   128
#define CDIM  128
#define NBLK  5
#define GRID  32768      // 32^3
#define BB    2
#define TT    100000
#define NTILES 12500     // B*T/16
#define TPB    6250      // tiles per batch

typedef __attribute__((ext_vector_type(16))) _Float16 v16h;
typedef __attribute__((ext_vector_type(8)))  _Float16 v8h;
typedef __attribute__((ext_vector_type(8)))  float    v8f;

// ---------------- workspace layout (bytes, all 256-aligned) ----------------
// Wp   @ 0         : packed f16 weights (851,968 B)
// net  @ 851968    : B*T*128 f16 (51,200,000 B)
// cm0  @ 52051968  : B*G*128 u32 (33,554,432 B)   [aliased by sums at the end]
// cm1  @ 85606400  : B*G*128 u32 (33,554,432 B)   [aliased by cnt at the end]
#define WS_NET  851968ULL
#define WS_CM0  52051968ULL
#define WS_CM1  85606400ULL

// packed weight offsets (in halves): per block [w0:32768][sc:32768][w1:16384]
#define WPK_BLK 81920
#define WPK_FCC 409600
#define WPK_TOT 425984

// block gemm persistent config
#define BG_WGS   250
#define BG_WAVES 8
#define BG_GWAVES (BG_WGS * BG_WAVES)                       // 2000
#define BG_SMEM_BYTES (WPK_BLK * 2 + BG_WAVES * 2048 * 2)   // 196608

// ---------- monotonic float <-> uint encoding for atomicMax on floats ------
__device__ __forceinline__ unsigned enc_f32(float v) {
  unsigned u = __float_as_uint(v);
  return (u & 0x80000000u) ? ~u : (u | 0x80000000u);
}
__device__ __forceinline__ float dec_f32(unsigned e) {
  unsigned u = (e & 0x80000000u) ? (e & 0x7fffffffu) : ~e;
  return __uint_as_float(u);
}

// ---------------------------------------------------------------------------
// Pack all weight matrices (K x 128, row-major f32) into per-lane WMMA
// B-fragment order, f16:  ((kb*8+nb)*32 + lane)*16 + e
// ---------------------------------------------------------------------------
__global__ void pack_weights_kernel(const float* __restrict__ fc0_w,
                                    const float* __restrict__ fc1_w,
                                    const float* __restrict__ sc_w,
                                    const float* __restrict__ fc_c_w,
                                    _Float16* __restrict__ Wp) {
  long e = (long)blockIdx.x * blockDim.x + threadIdx.x;
  if (e >= WPK_TOT) return;
  const float* src;
  long r, dstoff;
  if (e < (long)NBLK * WPK_BLK) {
    int blk = (int)(e / WPK_BLK);
    r = e % WPK_BLK;
    if (r < 32768)      { src = fc0_w + (size_t)blk * 32768; dstoff = (long)blk * WPK_BLK; }
    else if (r < 65536) { src = sc_w  + (size_t)blk * 32768; dstoff = (long)blk * WPK_BLK + 32768; r -= 32768; }
    else                { src = fc1_w + (size_t)blk * 16384; dstoff = (long)blk * WPK_BLK + 65536; r -= 65536; }
  } else {
    src = fc_c_w; dstoff = WPK_FCC; r = e - WPK_FCC;
  }
  long tile = r >> 9;            // /512 (32 lanes * 16 halves)
  int  rr   = (int)(r & 511);
  int  lane = rr >> 4;
  int  elem = rr & 15;
  int  nb   = (int)(tile & 7);   // N=128 -> 8 n-tiles
  int  kb   = (int)(tile >> 3);
  int  n = nb * 16 + (lane & 15);
  int  k = kb * 32 + (lane >> 4) * 8 + (elem & 7) + (elem >> 3) * 16;
  Wp[dstoff + r] = (_Float16)src[(size_t)k * 128 + n];
}

// ----- shared WMMA step: relu(A)@W0 into accH, A@Wsc into accS (one kb) ----
__device__ __forceinline__ void mma_kb(const v16h& araw, int kb,
                                       const _Float16* __restrict__ w0s,
                                       const _Float16* __restrict__ scs,
                                       int lane, v8f* accH, v8f* accS) {
  v16h arelu;
  const _Float16 zh = (_Float16)0.f;
#pragma unroll
  for (int j = 0; j < 16; ++j) arelu[j] = araw[j] > zh ? araw[j] : zh;
#pragma unroll
  for (int nb = 0; nb < 8; ++nb) {
    v16h bw = *(const v16h*)(w0s + (kb * 8 + nb) * 512 + lane * 16);
    accH[nb] = __builtin_amdgcn_wmma_f32_16x16x32_f16(
        false, arelu, false, bw, (short)0, accH[nb], false, false);
    v16h bs = *(const v16h*)(scs + (kb * 8 + nb) * 512 + lane * 16);
    accS[nb] = __builtin_amdgcn_wmma_f32_16x16x32_f16(
        false, araw, false, bs, (short)0, accS[nb], false, false);
  }
}

// ------------------- fully fused ResNet block (persistent WGs) -------------
// A-operand (16 x 256) built on the fly:
//   block 0  : positional linear  points(16x3) @ fc_pos_w  (3 FMAs/element)
//   blocks>0 : [ net(16x128) | decode(cellmax[g(row)]) gather ]
// Epilogue: net_out = X@Wsc + relu(relu(X)@W0+b0)@W1 + b1, stored f16 AND
//           atomicMax-scattered (monotone u32) into next stage's cellmax.
__global__ __launch_bounds__(256)
void block_gemm_kernel(const float* __restrict__ pts,      // block0 only
                       const float* __restrict__ pw,       // fc_pos_w (3,256)
                       const float* __restrict__ pb,       // fc_pos_b (256)
                       const _Float16* __restrict__ netin, // blocks>0
                       const unsigned* __restrict__ cm_in, // blocks>0 (null => block0)
                       const int* __restrict__ idx2,       // int64 as int pairs
                       const _Float16* __restrict__ Wp,    // [w0|sc|w1] packed
                       const float* __restrict__ b0,
                       const float* __restrict__ b1,
                       _Float16* __restrict__ netout,
                       unsigned* __restrict__ cm_out) {    // null => no scatter
  extern __shared__ _Float16 smem[];
  _Float16* w0s = smem;                  // 32768 halves
  _Float16* scs = smem + 32768;          // 32768 halves
  _Float16* w1s = smem + 65536;          // 16384 halves
  _Float16* bnc = smem + WPK_BLK;        // 8 waves * 2048 halves (H bounce)

  // ---- cooperative weight staging: 160 KB global -> LDS as b128 chunks ----
  {
    const uint4* src = (const uint4*)Wp;
    uint4*       dst = (uint4*)smem;
    for (int i = threadIdx.x; i < WPK_BLK * 2 / 16; i += 256) dst[i] = src[i];
  }
  __syncthreads();

  const int wave = threadIdx.x >> 5;
  const int lane = threadIdx.x & 31;
  const int m  = lane & 15;
  const int hi = lane >> 4;
  _Float16* hl = bnc + wave * 2048;
  const int gwave = blockIdx.x * BG_WAVES + wave;

  for (int tile = gwave; tile < NTILES; tile += BG_GWAVES) {
    const size_t row0 = (size_t)tile * 16;
    const int    bb   = tile / TPB;

    v8f zero = {0.f, 0.f, 0.f, 0.f, 0.f, 0.f, 0.f, 0.f};
    v8f accH[8], accS[8];
#pragma unroll
    for (int nb = 0; nb < 8; ++nb) { accH[nb] = zero; accS[nb] = zero; }

    // ---- GEMM pass 1: K = 256 (8 k-blocks of 32), A built on the fly ----
    if (cm_in == nullptr) {
      // block 0: positional linear on the fly (weights L2/const-cache hot)
      const float p0 = pts[(row0 + m) * 3 + 0];
      const float p1 = pts[(row0 + m) * 3 + 1];
      const float p2 = pts[(row0 + m) * 3 + 2];
      for (int kb = 0; kb < 8; ++kb) {
        const int k0 = kb * 32 + hi * 8;
        v16h araw;
#pragma unroll
        for (int j = 0; j < 8; ++j) {
          int ka = k0 + j, kc = k0 + 16 + j;
          araw[j]     = (_Float16)(pb[ka] + p0 * pw[ka] + p1 * pw[256 + ka] + p2 * pw[512 + ka]);
          araw[j + 8] = (_Float16)(pb[kc] + p0 * pw[kc] + p1 * pw[256 + kc] + p2 * pw[512 + kc]);
        }
        mma_kb(araw, kb, w0s, scs, lane, accH, accS);
      }
    } else {
      // blocks 1..4: first 128 K from net, second 128 K from pooled cellmax
      const int    grow  = idx2[2 * (row0 + m)];
      const size_t cbase = ((size_t)bb * GRID + grow) << 7;
      for (int kb = 0; kb < 4; ++kb) {
        const _Float16* ap = netin + (row0 + m) * 128 + kb * 32 + hi * 8;
        v8h a0 = *(const v8h*)ap;
        v8h a1 = *(const v8h*)(ap + 16);
        v16h araw;
#pragma unroll
        for (int j = 0; j < 8; ++j) { araw[j] = a0[j]; araw[j + 8] = a1[j]; }
        mma_kb(araw, kb, w0s, scs, lane, accH, accS);
      }
      for (int kb = 4; kb < 8; ++kb) {
        const unsigned* cp = cm_in + cbase + (kb - 4) * 32 + hi * 8;
        alignas(16) unsigned e[16];
        *(uint4*)(&e[0])  = *(const uint4*)(cp);
        *(uint4*)(&e[4])  = *(const uint4*)(cp + 4);
        *(uint4*)(&e[8])  = *(const uint4*)(cp + 16);
        *(uint4*)(&e[12]) = *(const uint4*)(cp + 20);
        v16h araw;
#pragma unroll
        for (int j = 0; j < 8; ++j) {
          araw[j]     = (_Float16)dec_f32(e[j]);
          araw[j + 8] = (_Float16)dec_f32(e[8 + j]);
        }
        mma_kb(araw, kb, w0s, scs, lane, accH, accS);
      }
    }

    // ---- relu(H + b0) -> per-wave LDS bounce (D-layout -> A-layout) ----
    // Intra-wave DS ops are in-order (DScnt); no barrier needed.
#pragma unroll
    for (int nb = 0; nb < 8; ++nb) {
#pragma unroll
      for (int v = 0; v < 8; ++v) {
        float h = accH[nb][v] + b0[nb * 16 + m];
        h = h > 0.f ? h : 0.f;
        hl[(v + hi * 8) * 128 + nb * 16 + m] = (_Float16)h;
      }
    }

    // ---- GEMM pass 2: K = 128, accumulate into skip-connection accS ----
    for (int kb = 0; kb < 4; ++kb) {
      const _Float16* ap = hl + m * 128 + kb * 32 + hi * 8;
      v8h a0 = *(const v8h*)ap;
      v8h a1 = *(const v8h*)(ap + 16);
      v16h a;
#pragma unroll
      for (int j = 0; j < 8; ++j) { a[j] = a0[j]; a[j + 8] = a1[j]; }
#pragma unroll
      for (int nb = 0; nb < 8; ++nb) {
        v16h bw = *(const v16h*)(w1s + (kb * 8 + nb) * 512 + lane * 16);
        accS[nb] = __builtin_amdgcn_wmma_f32_16x16x32_f16(
            false, a, false, bw, (short)0, accS[nb], false, false);
      }
    }

    // ---- + b1, store net f16, fused segment-max scatter for next stage ----
    int gm[8];
    if (cm_out) {
#pragma unroll
      for (int v = 0; v < 8; ++v) gm[v] = idx2[2 * (row0 + v + hi * 8)];
    }
#pragma unroll
    for (int nb = 0; nb < 8; ++nb) {
#pragma unroll
      for (int v = 0; v < 8; ++v) {
        float r = accS[nb][v] + b1[nb * 16 + m];
        netout[(row0 + v + hi * 8) * 128 + nb * 16 + m] = (_Float16)r;
        if (cm_out) {
          atomicMax(&cm_out[(((size_t)bb * GRID + gm[v]) << 7) + nb * 16 + m],
                    enc_f32(r));
        }
      }
    }
  }
}

// --------- final 128x128 projection (WMMA) + segment-sum scatter -----------
__global__ __launch_bounds__(128)
void final_gemm_kernel(const _Float16* __restrict__ net,
                       const _Float16* __restrict__ Cp,  // packed fc_c_w
                       const float* __restrict__ cb,
                       const int* __restrict__ idx2,
                       float* __restrict__ sums,
                       float* __restrict__ cnt) {
  const int wave = threadIdx.x >> 5;
  const int lane = threadIdx.x & 31;
  const int tile = blockIdx.x * 4 + wave;
  const int m  = lane & 15;
  const int hi = lane >> 4;
  const size_t row0 = (size_t)tile * 16;
  const int b = tile / TPB;

  v8f zero = {0.f, 0.f, 0.f, 0.f, 0.f, 0.f, 0.f, 0.f};
  v8f acc[8];
#pragma unroll
  for (int nb = 0; nb < 8; ++nb) acc[nb] = zero;

  for (int kb = 0; kb < 4; ++kb) {
    const _Float16* ap = net + (row0 + m) * 128 + kb * 32 + hi * 8;
    v8h a0 = *(const v8h*)ap;
    v8h a1 = *(const v8h*)(ap + 16);
    v16h a;
#pragma unroll
    for (int j = 0; j < 8; ++j) { a[j] = a0[j]; a[j + 8] = a1[j]; }
#pragma unroll
    for (int nb = 0; nb < 8; ++nb) {
      v16h bw = *(const v16h*)(Cp + (size_t)(kb * 8 + nb) * 512 + lane * 16);
      acc[nb] = __builtin_amdgcn_wmma_f32_16x16x32_f16(
          false, a, false, bw, (short)0, acc[nb], false, false);
    }
  }

  int gm[8];
#pragma unroll
  for (int v = 0; v < 8; ++v) gm[v] = idx2[2 * (row0 + v + hi * 8)];
  size_t sbase = (size_t)b * GRID * 128;
#pragma unroll
  for (int nb = 0; nb < 8; ++nb) {
#pragma unroll
    for (int v = 0; v < 8; ++v) {
      float r = acc[nb][v] + cb[nb * 16 + m];
      atomicAdd(&sums[sbase + ((size_t)gm[v] << 7) + nb * 16 + m], r);
    }
  }
  if (lane < 16) {
    int g = idx2[2 * (row0 + lane)];
    atomicAdd(&cnt[(size_t)b * GRID + g], 1.0f);
  }
}

// ----------------- mean + transpose to (B, CDIM, 32^3) ---------------------
__global__ void mean_kernel(const float* __restrict__ sums,
                            const float* __restrict__ cnt,
                            float* __restrict__ out) {
  size_t gid = (size_t)blockIdx.x * blockDim.x + threadIdx.x;
  if (gid >= (size_t)BB * CDIM * GRID) return;
  int g    = (int)(gid & (GRID - 1));
  int rest = (int)(gid >> 15);
  int c    = rest & 127;
  int b    = rest >> 7;
  float n  = cnt[(size_t)b * GRID + g];
  float v  = 0.f;
  if (n > 0.f) v = sums[(((size_t)b * GRID + g) << 7) + c] / n;
  out[gid] = v;
}

// ---------------------------------------------------------------------------
extern "C" void kernel_launch(void* const* d_in, const int* in_sizes, int n_in,
                              void* d_out, int out_size, void* d_ws, size_t ws_size,
                              hipStream_t stream) {
  const float* points   = (const float*)d_in[0];
  const int*   idx2     = (const int*)d_in[1];   // int64 -> low dword valid (g < 32768)
  const float* fc_pos_w = (const float*)d_in[2];
  const float* fc_pos_b = (const float*)d_in[3];
  const float* fc0_w    = (const float*)d_in[4];
  const float* fc0_b    = (const float*)d_in[5];
  const float* fc1_w    = (const float*)d_in[6];
  const float* fc1_b    = (const float*)d_in[7];
  const float* sc_w     = (const float*)d_in[8];
  const float* fc_c_w   = (const float*)d_in[9];
  const float* fc_c_b   = (const float*)d_in[10];

  char* ws = (char*)d_ws;
  _Float16* Wp  = (_Float16*)(ws);
  _Float16* net = (_Float16*)(ws + WS_NET);
  unsigned* cm0 = (unsigned*)(ws + WS_CM0);
  unsigned* cm1 = (unsigned*)(ws + WS_CM1);
  float*    sums = (float*)(ws + WS_CM0);   // alias: cm0 dead after block 3
  float*    cnt  = (float*)(ws + WS_CM1);   // alias: cm1 dead after block 4

  pack_weights_kernel<<<(WPK_TOT + 255) / 256, 256, 0, stream>>>(
      fc0_w, fc1_w, sc_w, fc_c_w, Wp);

  unsigned* cm[2] = {cm0, cm1};

  // block 0: positional linear fused, scatter -> cm0
  hipMemsetAsync(cm0, 0, 33554432ULL, stream);
  block_gemm_kernel<<<BG_WGS, 256, BG_SMEM_BYTES, stream>>>(
      points, fc_pos_w, fc_pos_b, nullptr, nullptr, idx2,
      Wp, fc0_b, fc1_b, net, cm0);

  // blocks 1..4: read cm[(i+1)&1], scatter -> cm[i&1] (none for block 4)
  for (int i = 1; i < NBLK; ++i) {
    unsigned* cin  = cm[(i + 1) & 1];
    unsigned* cout = (i < NBLK - 1) ? cm[i & 1] : nullptr;
    if (cout) hipMemsetAsync(cout, 0, 33554432ULL, stream);
    block_gemm_kernel<<<BG_WGS, 256, BG_SMEM_BYTES, stream>>>(
        nullptr, nullptr, nullptr, net, cin, idx2,
        Wp + (size_t)i * WPK_BLK, fc0_b + i * HID, fc1_b + i * HID, net, cout);
  }

  // final projection + segment mean (sums/cnt alias cm0/cm1 -- both dead now)
  hipMemsetAsync(sums, 0, 33554432ULL, stream);
  hipMemsetAsync(cnt, 0, 262144ULL, stream);
  final_gemm_kernel<<<3125, 128, 0, stream>>>(net, Wp + WPK_FCC, fc_c_b, idx2, sums, cnt);
  mean_kernel<<<32768, 256, 0, stream>>>(sums, cnt, (float*)d_out);
}
